// MemoryModel_77068893160300
// MI455X (gfx1250) — matrix operand order repaired and checked by reading, not compiled, not run on hardware
//
#include <hip/hip_runtime.h>
#include <hip/hip_bf16.h>
#include <stdint.h>

// Problem constants (match reference)
#define HD   64      // hidden dim
#define MM   512     // memory positions
#define LL   512     // sequence length
#define BB   1024    // batch
#define NTH  256     // threads per block (8 wave32)

typedef __attribute__((ext_vector_type(16))) __bf16          v16bf;
typedef __attribute__((ext_vector_type(16))) unsigned short  v16u;
typedef __attribute__((ext_vector_type(8)))  float           v8f;

__device__ __forceinline__ unsigned short f2bf(float f) {
    union { float f; unsigned u; } v; v.f = f;
    unsigned u = v.u;
    unsigned r = (u + 0x7FFFu + ((u >> 16) & 1u)) >> 16;   // round-to-nearest-even
    return (unsigned short)r;
}
__device__ __forceinline__ float bflo(unsigned w2) {       // low bf16 of a dword
    return __uint_as_float(w2 << 16);
}
__device__ __forceinline__ float bfhi(unsigned w2) {       // high bf16 of a dword
    return __uint_as_float(w2 & 0xFFFF0000u);
}

// ---- LDS layout (bytes) --------------------------------------------------
// All bf16 arrays are dword-packed along the reduction dim:
//   WihT : pairs (j,j+1) per output o  -> dword idx (j/2)*192 + o
//   WhhT : dword idx (j/2)*192 + o
//   WqT  : dword idx (j/2)*64  + o
//   Kt   : pairs (h,h+1) per m         -> dword idx (h/2)*512 + m
//   Vv   : pairs (m,m+1) per h         -> dword idx (m/2)*64  + h
#define OF_WIHT 0                    // 128*192*2 = 49152
#define OF_WHHT 49152                //  64*192*2 = 24576
#define OF_WQT  73728                //   64*64*2 =  8192
#define OF_BIH  81920                // 192 f32
#define OF_BHH  82688                // 192 f32
#define OF_BQ   83456                // 64  f32
#define OF_KT   83712                // 64*512*2 = 65536
#define OF_VV   149248               // 512*64*2 = 65536
#define OF_INP  214784               // inp[2H] f32  (x | retrieved)
#define OF_HV   215296               // h[H]  f32
#define OF_HN   215552               // h_new[H] f32
#define OF_GI   215808               // gi[3H] f32
#define OF_GH   216576               // gh[3H] f32
#define OF_QV   217344               // q[H] f32 (scale folded in)
#define OF_SC   217600               // scores[M] f32 : 2048
#define OF_RED  219648               // reduction scratch (64+8+1 f32)
#define SMEM_BYTES 219968

__global__ __launch_bounds__(NTH, 1)
void memgru_fused_kernel(
    const int*   __restrict__ seq,     const float* __restrict__ memory,
    const float* __restrict__ embed_W,
    const float* __restrict__ W_ih,    const float* __restrict__ W_hh,
    const float* __restrict__ b_ih,    const float* __restrict__ b_hh,
    const float* __restrict__ Wq,      const float* __restrict__ bq,
    const float* __restrict__ Wk,      const float* __restrict__ bk,
    const float* __restrict__ Wv,      const float* __restrict__ bv,
    const float* __restrict__ W_head,  const float* __restrict__ b_head,
    float* __restrict__ out, int out_size)
{
    extern __shared__ char smem[];
    unsigned short* WihT = (unsigned short*)(smem + OF_WIHT);
    unsigned short* WhhT = (unsigned short*)(smem + OF_WHHT);
    unsigned short* WqT  = (unsigned short*)(smem + OF_WQT);
    float* bihs = (float*)(smem + OF_BIH);
    float* bhhs = (float*)(smem + OF_BHH);
    float* bqs  = (float*)(smem + OF_BQ);
    unsigned short* Kt = (unsigned short*)(smem + OF_KT);
    unsigned short* Vv = (unsigned short*)(smem + OF_VV);
    float* inp = (float*)(smem + OF_INP);
    float* hv  = (float*)(smem + OF_HV);
    float* hn  = (float*)(smem + OF_HN);
    float* gi  = (float*)(smem + OF_GI);
    float* gh  = (float*)(smem + OF_GH);
    float* qv  = (float*)(smem + OF_QV);
    float* sc  = (float*)(smem + OF_SC);
    float* red = (float*)(smem + OF_RED);

    const unsigned* Wihp = (const unsigned*)WihT;
    const unsigned* Whhp = (const unsigned*)WhhT;
    const unsigned* Wqp  = (const unsigned*)WqT;
    const unsigned* Ktp  = (const unsigned*)Kt;
    const unsigned* Vvp  = (const unsigned*)Vv;

    const int tid  = threadIdx.x;
    const int wave = tid >> 5;
    const int lane = tid & 31;
    const int b    = blockIdx.x;

    // ---- Phase 0: cache weights (bf16, transposed+pair-packed) -----------
    for (int i = tid; i < 192 * 128; i += NTH) {
        int o = i / 128, j = i % 128;
        WihT[(j >> 1) * 384 + o * 2 + (j & 1)] = f2bf(W_ih[o * 128 + j]);
    }
    for (int i = tid; i < 192 * 64; i += NTH) {
        int o = i / 64, j = i % 64;
        WhhT[(j >> 1) * 384 + o * 2 + (j & 1)] = f2bf(W_hh[o * 64 + j]);
    }
    for (int i = tid; i < 64 * 64; i += NTH) {
        int o = i / 64, j = i % 64;
        WqT[(j >> 1) * 128 + o * 2 + (j & 1)] = f2bf(Wq[o * 64 + j]);
    }
    if (tid < 192) { bihs[tid] = b_ih[tid]; bhhs[tid] = b_hh[tid]; }
    if (tid < HD)  { bqs[tid] = bq[tid]; hv[tid] = 0.0f; inp[HD + tid] = 0.0f; }
    __syncthreads();

    // ---- Phase 1: K/V projection via WMMA bf16 ----------------------------
    // k_mem[m,h] = memory[b,m,:] . Wk[h,:] + bk[h]  (same for V with Wv/bv).
    // A = 16 memory rows x 32 k ; B = weight tile (shared operand) ; D = f32 16x16.
    const size_t memBase = (size_t)b * (MM * HD);
    const int lrow   = lane & 15;
    const int hiHalf = lane >> 4;

    // --- K projection ---
    for (int mt = wave; mt < MM / 16; mt += 8) {
        const int mrow = mt * 16 + lrow;
        const float* mrowp = memory + memBase + (size_t)mrow * HD;
        if (mt + 8 < MM / 16)
            __builtin_prefetch(memory + memBase + (size_t)(mrow + 128) * HD, 0, 0);
        v16u a0u, a1u;
        #pragma unroll
        for (int g = 0; g < 8; ++g) {
            int kb = ((g < 4) ? 2 * g : 16 + 2 * (g - 4)) + (hiHalf ? 8 : 0);
            a0u[2 * g]     = f2bf(mrowp[kb]);
            a0u[2 * g + 1] = f2bf(mrowp[kb + 1]);
            a1u[2 * g]     = f2bf(mrowp[32 + kb]);
            a1u[2 * g + 1] = f2bf(mrowp[32 + kb + 1]);
        }
        v16bf a0 = __builtin_bit_cast(v16bf, a0u);
        v16bf a1 = __builtin_bit_cast(v16bf, a1u);
        #pragma unroll
        for (int nt = 0; nt < HD / 16; ++nt) {
            const int n = nt * 16 + lrow;
            const float* wrow = Wk + n * HD + (hiHalf ? 8 : 0);
            v16u b0u, b1u;
            #pragma unroll
            for (int g = 0; g < 16; ++g) { b0u[g] = f2bf(wrow[(g & 7) + ((g >> 3) << 4)]); b1u[g] = f2bf(wrow[32 + (g & 7) + ((g >> 3) << 4)]); }
            v16bf bb0 = __builtin_bit_cast(v16bf, b0u);
            v16bf bb1 = __builtin_bit_cast(v16bf, b1u);
            v8f acc = {};
            acc = __builtin_amdgcn_wmma_f32_16x16x32_bf16(false, a0, false, bb0, (short)0, acc, false, false);
            acc = __builtin_amdgcn_wmma_f32_16x16x32_bf16(false, a1, false, bb1, (short)0, acc, false, false);
            const int hcol = nt * 16 + lrow;             // D: col = lane%16
            const float bb = bk[hcol];
            #pragma unroll
            for (int r = 0; r < 8; ++r) {
                int mout = mt * 16 + r + (hiHalf ? 8 : 0);  // D: row = r + 8*hiHalf
                Kt[(hcol >> 1) * 1024 + mout * 2 + (hcol & 1)] = f2bf(acc[r] + bb);
            }
        }
    }
    // --- V projection ---
    for (int mt = wave; mt < MM / 16; mt += 8) {
        const int mrow = mt * 16 + lrow;
        const float* mrowp = memory + memBase + (size_t)mrow * HD;
        v16u a0u, a1u;
        #pragma unroll
        for (int g = 0; g < 8; ++g) {
            int kb = ((g < 4) ? 2 * g : 16 + 2 * (g - 4)) + (hiHalf ? 8 : 0);
            a0u[2 * g]     = f2bf(mrowp[kb]);
            a0u[2 * g + 1] = f2bf(mrowp[kb + 1]);
            a1u[2 * g]     = f2bf(mrowp[32 + kb]);
            a1u[2 * g + 1] = f2bf(mrowp[32 + kb + 1]);
        }
        v16bf a0 = __builtin_bit_cast(v16bf, a0u);
        v16bf a1 = __builtin_bit_cast(v16bf, a1u);
        #pragma unroll
        for (int nt = 0; nt < HD / 16; ++nt) {
            const int n = nt * 16 + lrow;
            const float* wrow = Wv + n * HD + (hiHalf ? 8 : 0);
            v16u b0u, b1u;
            #pragma unroll
            for (int g = 0; g < 16; ++g) { b0u[g] = f2bf(wrow[(g & 7) + ((g >> 3) << 4)]); b1u[g] = f2bf(wrow[32 + (g & 7) + ((g >> 3) << 4)]); }
            v16bf bb0 = __builtin_bit_cast(v16bf, b0u);
            v16bf bb1 = __builtin_bit_cast(v16bf, b1u);
            v8f acc = {};
            acc = __builtin_amdgcn_wmma_f32_16x16x32_bf16(false, a0, false, bb0, (short)0, acc, false, false);
            acc = __builtin_amdgcn_wmma_f32_16x16x32_bf16(false, a1, false, bb1, (short)0, acc, false, false);
            const int hcol = nt * 16 + lrow;
            const float bb = bv[hcol];
            #pragma unroll
            for (int r = 0; r < 8; ++r) {
                int mout = mt * 16 + r + (hiHalf ? 8 : 0);
                Vv[(mout >> 1) * 128 + hcol * 2 + (mout & 1)] = f2bf(acc[r] + bb);
            }
        }
    }
    __syncthreads();

    // ---- Phase 2: 512-step GRU + attention recurrence (all from LDS) ------
    const int* seqRow = seq + (size_t)b * LL;
    for (int t = 0; t < LL; ++t) {
        if (tid < HD) {
            int tok = seqRow[t];
            inp[tid] = embed_W[tok * HD + tid];
        }
        __syncthreads();
        // gates: gi = W_ih @ [x;r] + b_ih ; gh = W_hh @ h + b_hh
        if (tid < 192) {
            float ai = bihs[tid];
            #pragma unroll 4
            for (int j2 = 0; j2 < HD; ++j2) {             // 64 dword pairs = 128 j
                unsigned w2 = Wihp[j2 * 192 + tid];
                ai += bflo(w2) * inp[2 * j2] + bfhi(w2) * inp[2 * j2 + 1];
            }
            gi[tid] = ai;
            float ah = bhhs[tid];
            #pragma unroll 4
            for (int j2 = 0; j2 < HD / 2; ++j2) {         // 32 dword pairs = 64 j
                unsigned w2 = Whhp[j2 * 192 + tid];
                ah += bflo(w2) * hv[2 * j2] + bfhi(w2) * hv[2 * j2 + 1];
            }
            gh[tid] = ah;
        }
        __syncthreads();
        // GRU cell
        if (tid < HD) {
            float rg = 1.0f / (1.0f + __expf(-(gi[tid] + gh[tid])));
            float z  = 1.0f / (1.0f + __expf(-(gi[HD + tid] + gh[HD + tid])));
            float nn = tanhf(gi[2 * HD + tid] + rg * gh[2 * HD + tid]);
            hn[tid] = (1.0f - z) * nn + z * hv[tid];
        }
        __syncthreads();
        // q = (Wq @ h_new + bq) / sqrt(H)  (scale folded); commit h
        if (tid < HD) {
            float a = bqs[tid];
            #pragma unroll 4
            for (int j2 = 0; j2 < HD / 2; ++j2) {
                unsigned w2 = Wqp[j2 * 64 + tid];
                a += bflo(w2) * hn[2 * j2] + bfhi(w2) * hn[2 * j2 + 1];
            }
            qv[tid] = a * 0.125f;
            hv[tid] = hn[tid];
        }
        __syncthreads();
        // scores[m] = q . K[m]   (Kt pair-packed: dword = (h,h+1) at col m)
        for (int m = tid; m < MM; m += NTH) {
            float s = 0.0f;
            #pragma unroll 4
            for (int h2 = 0; h2 < HD / 2; ++h2) {
                unsigned k2 = Ktp[h2 * MM + m];
                s += bflo(k2) * qv[2 * h2] + bfhi(k2) * qv[2 * h2 + 1];
            }
            sc[m] = s;
        }
        if (tid >= 192) inp[HD + (tid - 192)] = 0.0f;     // clear retrieved accum
        __syncthreads();
        // softmax: max
        if (tid < 64) {
            float mx = sc[tid * 8];
            #pragma unroll
            for (int i = 1; i < 8; ++i) mx = fmaxf(mx, sc[tid * 8 + i]);
            red[tid] = mx;
        }
        __syncthreads();
        if (tid < 8) {
            float mx = red[tid * 8];
            #pragma unroll
            for (int i = 1; i < 8; ++i) mx = fmaxf(mx, red[tid * 8 + i]);
            red[64 + tid] = mx;
        }
        __syncthreads();
        if (tid == 0) {
            float mx = red[64];
            #pragma unroll
            for (int i = 1; i < 8; ++i) mx = fmaxf(mx, red[64 + i]);
            red[72] = mx;
        }
        __syncthreads();
        const float mx = red[72];
        for (int m = tid; m < MM; m += NTH) sc[m] = __expf(sc[m] - mx);
        __syncthreads();
        // softmax: sum
        if (tid < 64) {
            float s = 0.0f;
            #pragma unroll
            for (int i = 0; i < 8; ++i) s += sc[tid * 8 + i];
            red[tid] = s;
        }
        __syncthreads();
        if (tid < 8) {
            float s = 0.0f;
            #pragma unroll
            for (int i = 0; i < 8; ++i) s += red[tid * 8 + i];
            red[64 + tid] = s;
        }
        __syncthreads();
        if (tid == 0) {
            float s = 0.0f;
            #pragma unroll
            for (int i = 0; i < 8; ++i) s += red[64 + i];
            red[72] = 1.0f / s;
        }
        __syncthreads();
        const float inv = red[72];
        for (int m = tid; m < MM; m += NTH) sc[m] *= inv;
        __syncthreads();
        // retrieved = w @ V  (Vv pair-packed: dword = (m,m+1) at col h)
        {
            const int hh = tid & 63;
            const int c0 = (tid >> 6) * 64;               // 64 dword pairs = 128 m
            float acc = 0.0f;
            #pragma unroll 4
            for (int m2 = c0; m2 < c0 + 64; ++m2) {
                unsigned v2 = Vvp[m2 * HD + hh];
                acc += bflo(v2) * sc[2 * m2] + bfhi(v2) * sc[2 * m2 + 1];
            }
            atomicAdd(&inp[HD + hh], acc);                // ds_add_f32
        }
        __syncthreads();
    }

    // ---- head: logits = h @ W_head^T + b_head ----------------------------
    if (tid < 64) {
        float acc = b_head[tid];
        #pragma unroll 4
        for (int h = 0; h < HD; ++h) acc += W_head[tid * HD + h] * hv[h];
        out[(size_t)b * 64 + tid] = acc;
    }
    if (b == 0 && tid == 0 && out_size > BB * 64) out[BB * 64] = 0.0f;  // aux scalar
}

extern "C" void kernel_launch(void* const* d_in, const int* in_sizes, int n_in,
                              void* d_out, int out_size, void* d_ws, size_t ws_size,
                              hipStream_t stream) {
    (void)in_sizes; (void)n_in; (void)d_ws; (void)ws_size;
    const int*   seq    = (const int*)  d_in[0];
    const float* memory = (const float*)d_in[1];
    const float* embedW = (const float*)d_in[2];
    const float* W_ih   = (const float*)d_in[3];
    const float* W_hh   = (const float*)d_in[4];
    const float* b_ih   = (const float*)d_in[5];
    const float* b_hh   = (const float*)d_in[6];
    const float* Wq     = (const float*)d_in[7];
    const float* bq     = (const float*)d_in[8];
    const float* Wk     = (const float*)d_in[9];
    const float* bk     = (const float*)d_in[10];
    const float* Wv     = (const float*)d_in[11];
    const float* bv     = (const float*)d_in[12];
    const float* W_head = (const float*)d_in[13];
    const float* b_head = (const float*)d_in[14];
    float* out = (float*)d_out;

    memgru_fused_kernel<<<BB, NTH, SMEM_BYTES, stream>>>(
        seq, memory, embedW, W_ih, W_hh, b_ih, b_hh,
        Wq, bq, Wk, bk, Wv, bv, W_head, b_head, out, out_size);
}
